// Model_24945170055688
// MI455X (gfx1250) — matrix operand order repaired and checked by reading, not compiled, not run on hardware
//
#include <hip/hip_runtime.h>
#include <hip/hip_bf16.h>

#define B_SZ   8192
#define DNUM   96
#define DCAT   160
#define DIN    256
#define NE     32
#define DB     512
#define DOUTN  10
#define DOUTP  16       // Wout padded to one full WMMA column tile
#define MT     128      // batch rows per workgroup
#define LDH    520      // padded LDS row stride (elements); 1040 B = 65*16, 16B aligned, bank-skewed

typedef __attribute__((ext_vector_type(16))) __bf16 bf16x16;
typedef __attribute__((ext_vector_type(8)))  float  floatx8;

union FragBF { bf16x16 v; int4 q[2]; };

// ---------------- prep kernels ----------------

__global__ void k_convert_x(const float* __restrict__ xn, const float* __restrict__ xc,
                            __bf16* __restrict__ xb) {
    int i = blockIdx.x * 256 + threadIdx.x;       // over B*DIN (exact multiple)
    int b = i >> 8, k = i & 255;
    float v = (k < DNUM) ? xn[b * DNUM + k] : xc[b * DCAT + (k - DNUM)];
    xb[i] = (__bf16)v;
}

// w: [E][K][N] fp32  ->  wt: [E][N][K] bf16
__global__ void k_transpose_w(const float* __restrict__ w, __bf16* __restrict__ wt,
                              int K, int N) {
    int i = blockIdx.x * 256 + threadIdx.x;       // over E*K*N (exact multiple)
    int n = i % N;
    int t = i / N;
    int k = t % K;
    int e = t / K;
    wt[((size_t)e * N + n) * K + k] = (__bf16)w[i];
}

// Wout: [E][DB][10] fp32 -> Woutt: [E][16][DB] bf16, zero-padded columns 10..15
__global__ void k_pad_wout(const float* __restrict__ w, __bf16* __restrict__ wt) {
    int i = blockIdx.x * 256 + threadIdx.x;       // over E*16*DB (exact multiple)
    int k = i % DB;
    int t = i / DB;
    int n = t % DOUTP;
    int e = t / DOUTP;
    float v = (n < DOUTN) ? w[((size_t)e * DB + k) * DOUTN + n] : 0.f;
    wt[i] = (__bf16)v;
}

// probs = softmax(x @ Wg + bg) in fp32; one wave per batch row, lane = expert
__global__ void k_gate(const float* __restrict__ xn, const float* __restrict__ xc,
                       const float* __restrict__ Wg, const float* __restrict__ bg,
                       float* __restrict__ probs) {
    int wave = threadIdx.x >> 5, lane = threadIdx.x & 31;
    int b = blockIdx.x * 8 + wave;
    float acc = bg[lane];
    for (int k = 0; k < DIN; ++k) {
        float xv = (k < DNUM) ? xn[b * DNUM + k] : xc[b * DCAT + (k - DNUM)];
        acc += xv * Wg[k * NE + lane];           // lanes read consecutive floats: coalesced
    }
    float m = acc;
    for (int off = 16; off; off >>= 1) m = fmaxf(m, __shfl_xor(m, off, 32));
    float p = __expf(acc - m);
    float s = p;
    for (int off = 16; off; off >>= 1) s += __shfl_xor(s, off, 32);
    probs[b * NE + lane] = p / s;
}

// ---------------- fused expert MLP ----------------
//
// One 16-row strip per wave. sIn: [MT][LDH] bf16, wt: [DB][KDIM] bf16 (expert base),
// output relu(x@W + b) -> sOut strip. A-frag layout (16-bit A 16x32):
//   lanes 0-15 : K 0-7 (bytes 0..15) and K 16-23 (bytes 32..47)
//   lanes 16-31: K 8-15 (bytes 16..31) and K 24-31 (bytes 48..63)
// B-frag layout (16-bit B 32x16), column n = lane&15, from K-contiguous wt row:
//   lanes 0-15 : K 0-7 and K 16-23, lanes 16-31: K 8-15 and K 24-31 (same as A)
template <int KDIM>
__device__ __forceinline__ void mlp_layer(const __bf16* __restrict__ sIn,
                                          const __bf16* __restrict__ wt,
                                          const float* __restrict__ bias,
                                          __bf16* __restrict__ sOut,
                                          int wave, int lane) {
    const int half = lane >> 4;        // 0 or 1
    const int lan  = lane & 15;

    // preload all A fragments for this strip's full K into registers
    const __bf16* aRow = sIn + (size_t)(wave * 16 + lan) * LDH + half * 8;
    FragBF a[KDIM / 32];
#pragma unroll
    for (int kt = 0; kt < KDIM / 32; ++kt) {
        a[kt].q[0] = *(const int4*)(aRow + kt * 32);
        a[kt].q[1] = *(const int4*)(aRow + kt * 32 + 16);
    }

    const int outRow = wave * 16 + half * 8;     // + r inside the store loop
    for (int nt = 0; nt < DB / 16; ++nt) {
        const int n0 = nt * 16;
        const __bf16* bRow = wt + (size_t)(n0 + lan) * KDIM + half * 8;
        floatx8 acc = {0.f, 0.f, 0.f, 0.f, 0.f, 0.f, 0.f, 0.f};
#pragma unroll
        for (int kt = 0; kt < KDIM / 32; ++kt) {
            FragBF bf;
            bf.q[0] = *(const int4*)(bRow + kt * 32);
            bf.q[1] = *(const int4*)(bRow + kt * 32 + 16);
            acc = __builtin_amdgcn_wmma_f32_16x16x32_bf16(
                false, a[kt].v, false, bf.v, (short)0, acc, false, false);
        }
        const float bv = bias[n0 + lan];
        __bf16* op = sOut + (size_t)outRow * LDH + n0 + lan;
#pragma unroll
        for (int r = 0; r < 8; ++r) {
            float v = acc[r] + bv;
            op[r * LDH] = (__bf16)fmaxf(v, 0.f);
        }
    }
}

__global__ void __launch_bounds__(256)
k_experts(const __bf16* __restrict__ xb, const __bf16* __restrict__ W1t,
          const __bf16* __restrict__ W2t, const float* __restrict__ b1,
          const float* __restrict__ b2, const __bf16* __restrict__ Woutt,
          const float* __restrict__ bout, float* __restrict__ y_all) {
    extern __shared__ char smem[];
    __bf16* sA = (__bf16*)smem;                                  // x tile, later h2
    __bf16* sH = (__bf16*)(smem + (size_t)MT * LDH * 2);         // h1

    const int e    = blockIdx.y;
    const int m0   = blockIdx.x * MT;
    const int wave = threadIdx.x >> 5;
    const int lane = threadIdx.x & 31;
    const int half = lane >> 4;
    const int lan  = lane & 15;

    // cooperative bf16 x-tile load: 128 rows x 256 cols = 4096 int4
    const __bf16* xg = xb + (size_t)m0 * DIN;
#pragma unroll
    for (int i = 0; i < 16; ++i) {
        int j = threadIdx.x + i * 256;   // 0..4095
        int r = j >> 5;                  // row (32 int4 per row)
        int c = j & 31;                  // int4 within row
        *(int4*)(sA + (size_t)r * LDH + c * 8) = *((const int4*)(xg + (size_t)r * DIN) + c);
    }
    __syncthreads();

    // layer 1: h1 = relu(x @ W1[e] + b1[e])        (K=256)
    mlp_layer<DIN>(sA, W1t + (size_t)e * DB * DIN, b1 + e * DB, sH, wave, lane);
    // layer 2: h2 = relu(h1 @ W2[e] + b2[e])       (K=512); each wave owns its rows -> no barrier
    mlp_layer<DB>(sH, W2t + (size_t)e * DB * DB, b2 + e * DB, sA, wave, lane);

    // head via WMMA: y = h2 @ Woutt[e] + bout[e], one 16-col tile (cols 10..15 are zero pad)
    {
        const __bf16* aRow = sA + (size_t)(wave * 16 + lan) * LDH + half * 8;
        const __bf16* bRow = Woutt + ((size_t)e * DOUTP + lan) * DB + half * 8;
        floatx8 acc = {0.f, 0.f, 0.f, 0.f, 0.f, 0.f, 0.f, 0.f};
#pragma unroll
        for (int kt = 0; kt < DB / 32; ++kt) {
            FragBF af, bf;
            af.q[0] = *(const int4*)(aRow + kt * 32);
            af.q[1] = *(const int4*)(aRow + kt * 32 + 16);
            bf.q[0] = *(const int4*)(bRow + kt * 32);
            bf.q[1] = *(const int4*)(bRow + kt * 32 + 16);
            acc = __builtin_amdgcn_wmma_f32_16x16x32_bf16(
                false, af.v, false, bf.v, (short)0, acc, false, false);
        }
        if (lan < DOUTN) {
            const float bv = bout[e * DOUTN + lan];
            const int rowBase = m0 + wave * 16 + half * 8;
#pragma unroll
            for (int r = 0; r < 8; ++r)
                y_all[((size_t)(rowBase + r) * NE + e) * DOUTN + lan] = acc[r] + bv;
        }
    }
}

// out[b,o] = sum_e probs[b,e] * y_all[b,e,o]   (deterministic reduction, no atomics)
__global__ void k_combine(const float* __restrict__ probs, const float* __restrict__ y_all,
                          float* __restrict__ out) {
    int i = blockIdx.x * 256 + threadIdx.x;
    if (i >= B_SZ * DOUTN) return;
    int b = i / DOUTN, o = i % DOUTN;
    float s = 0.f;
#pragma unroll
    for (int e = 0; e < NE; ++e)
        s += probs[b * NE + e] * y_all[((size_t)b * NE + e) * DOUTN + o];
    out[i] = s;
}

// ---------------- host ----------------

extern "C" void kernel_launch(void* const* d_in, const int* in_sizes, int n_in,
                              void* d_out, int out_size, void* d_ws, size_t ws_size,
                              hipStream_t stream) {
    const float* xn   = (const float*)d_in[0];
    const float* xc   = (const float*)d_in[1];
    const float* Wg   = (const float*)d_in[2];
    const float* bg   = (const float*)d_in[3];
    const float* W1   = (const float*)d_in[4];
    const float* b1   = (const float*)d_in[5];
    const float* W2   = (const float*)d_in[6];
    const float* b2   = (const float*)d_in[7];
    const float* Wout = (const float*)d_in[8];
    const float* bout = (const float*)d_in[9];
    float* out = (float*)d_out;

    char* ws = (char*)d_ws;
    __bf16* xb    = (__bf16*)(ws);                               //  4 MiB  (B*DIN bf16)
    __bf16* W1t   = (__bf16*)(ws + (4u  << 20));                 //  8 MiB  (E*DB*DIN bf16)
    __bf16* W2t   = (__bf16*)(ws + (12u << 20));                 // 16 MiB  (E*DB*DB bf16)
    float*  probs = (float*)(ws + (28u << 20));                  //  1 MiB  (B*E f32)
    float*  y_all = (float*)(ws + (29u << 20));                  // 10 MiB  (B*E*10 f32)
    __bf16* Woutt = (__bf16*)(ws + (39u << 20));                 // 0.5 MiB (E*16*DB bf16)

    k_convert_x<<<(B_SZ * DIN) / 256, 256, 0, stream>>>(xn, xc, xb);
    k_transpose_w<<<(NE * DIN * DB) / 256, 256, 0, stream>>>(W1, W1t, DIN, DB);
    k_transpose_w<<<(NE * DB * DB) / 256, 256, 0, stream>>>(W2, W2t, DB, DB);
    k_pad_wout<<<(NE * DOUTP * DB) / 256, 256, 0, stream>>>(Wout, Woutt);
    k_gate<<<B_SZ / 8, 256, 0, stream>>>(xn, xc, Wg, bg, probs);

    dim3 grid(B_SZ / MT, NE);
    size_t smem = 2u * MT * LDH * sizeof(__bf16);                // 266,240 B < 320 KB/WGP
    k_experts<<<grid, 256, smem, stream>>>(xb, W1t, W2t, b1, b2, Woutt, bout, y_all);

    k_combine<<<(B_SZ * DOUTN + 255) / 256, 256, 0, stream>>>(probs, y_all, out);
}
